// MultiScaleResidualQuantizer_10428180595131
// MI455X (gfx1250) — compile-verified, matrix operand order checked by reading, not run on hardware
//
#include <hip/hip_runtime.h>
#include <math.h>

// MI455X / gfx1250 (CDNA5, wave32). Codebook cosine search uses
// v_wmma_f32_16x16x32_f16 (K=32 == embedding dim, one WMMA per 16x16 tile).
// Everything else is fp32 VALU with LDS tiling; all reductions deterministic.

typedef __attribute__((ext_vector_type(16))) _Float16 v16h;
typedef __attribute__((ext_vector_type(8)))  float    v8f;

#define B_    16
#define C_    32
#define H_    32
#define NE_   4096
#define NELEM (B_*C_*H_*H_)          // 524288

// d_out layout (floats): f_hat | loss | perplexity | enc | idx
#define LOSS_OFF 524288
#define PERP_OFF 524289
#define ENC_OFF  524290
#define IDX_OFF  (524290 + 67108864)

__device__ __forceinline__ float cubw(float x) {   // bicubic kernel, a = -0.75
  x = fabsf(x);
  if (x <= 1.f) return (1.25f*x - 2.25f)*x*x + 1.f;
  if (x <  2.f) return ((-0.75f*x + 3.75f)*x - 6.f)*x + 3.f;
  return 0.f;
}

// ---------------- init: f_hat=0, f_rest=f, loss=0, counts=0 ----------------
__global__ void k_init(const float* __restrict__ f, float* __restrict__ out,
                       float* __restrict__ frest, int* __restrict__ counts) {
  int e = blockIdx.x * 256 + threadIdx.x;
  out[e]   = 0.f;          // f_hat region
  frest[e] = f[e];
  if (e < NE_) counts[e] = 0;
  if (e == 0)  out[LOSS_OFF] = 0.f;
}

__global__ void k_zero_enc(float* __restrict__ enc) {
  size_t i = ((size_t)blockIdx.x * 256 + threadIdx.x) * 4;
  enc[i] = 0.f; enc[i+1] = 0.f; enc[i+2] = 0.f; enc[i+3] = 0.f;
}

// ---------------- L2-normalize codebook rows -> f16 ----------------
__global__ void k_normw(const float* __restrict__ W, _Float16* __restrict__ Wn) {
  int row = blockIdx.x * 8 + (threadIdx.x >> 5);
  int c   = threadIdx.x & 31;            // lane == channel (wave32)
  float v = W[row * C_ + c];
  float ss = v * v;
#pragma unroll
  for (int m = 16; m; m >>= 1) ss += __shfl_xor(ss, m, 32);
  float dn = fmaxf(sqrtf(ss), 1e-12f);
  Wn[row * C_ + c] = (_Float16)(v / dn);
}

// ------- area-pool residual to pn x pn, L2-normalize rows -> f16 [R][32] -------
__global__ void k_prep(const float* __restrict__ frest, _Float16* __restrict__ rn,
                       int pn, int R) {
  int row = blockIdx.x * 8 + (threadIdx.x >> 5);
  int c   = threadIdx.x & 31;
  if (row >= R) return;                   // wave-uniform
  int pp  = pn * pn;
  int b   = row / pp;
  int rem = row - b * pp;
  int y   = rem / pn;
  int x   = rem - y * pn;
  int k   = H_ / pn;
  const float* base = frest + (((size_t)b * C_ + c) * H_ + y * k) * H_ + x * k;
  float acc = 0.f;
  for (int dy = 0; dy < k; ++dy)
    for (int dx = 0; dx < k; ++dx)
      acc += base[dy * H_ + dx];
  float v = acc / (float)(k * k);
  float ss = v * v;
#pragma unroll
  for (int m = 16; m; m >>= 1) ss += __shfl_xor(ss, m, 32);
  float dn = fmaxf(sqrtf(ss), 1e-12f);
  rn[(size_t)row * C_ + c] = (_Float16)(v / dn);
}

// ------- WMMA codebook search: argmax_n  rn[m,:] . Wn[n,:]  (K = 32) -------
__global__ void __launch_bounds__(128)
k_search(const _Float16* __restrict__ rn, const _Float16* __restrict__ Wn,
         int* __restrict__ idx, int ntiles) {
  int tile = blockIdx.x * 4 + (threadIdx.x >> 5);
  if (tile >= ntiles) return;             // wave-uniform: EXEC all-1s for WMMA
  int lane = threadIdx.x & 31;
  int r0   = tile * 16;
  int koff = (lane >> 4) * 8;             // K split per ISA 16-bit A layout

  // A: 16x32 f16, lane holds row M = lane&15; halves 0..7 -> K=koff+h,
  // halves 8..15 -> K=16+koff+h.
  const _Float16* ap = rn + (size_t)(r0 + (lane & 15)) * C_ + koff;
  v16h A;
#pragma unroll
  for (int h = 0; h < 8; ++h) { A[h] = ap[h]; A[8 + h] = ap[16 + h]; }

  float best[8];
  int   bestn[8];
#pragma unroll
  for (int v = 0; v < 8; ++v) { best[v] = -3.0e38f; bestn[v] = 0; }

  const _Float16* bp = Wn + (size_t)(lane & 15) * C_ + koff;
  for (int n0 = 0; n0 < NE_; n0 += 16) {
    const _Float16* q = bp + (size_t)n0 * C_;
    __builtin_prefetch(q + 16 * C_, 0, 0);     // global_prefetch_b8, next tile
    v16h Bm;                                   // B: mirrored layout, lane = N
#pragma unroll
    for (int h = 0; h < 8; ++h) { Bm[h] = q[h]; Bm[8 + h] = q[16 + h]; }
    v8f cz = {};
    v8f D = __builtin_amdgcn_wmma_f32_16x16x32_f16(
        false, A, false, Bm, (short)0, cz, false, false);
    int cand = n0 + (lane & 15);
    // D layout: lane holds N = lane&15; VGPR v holds row M = v + 8*(lane>=16)
#pragma unroll
    for (int v = 0; v < 8; ++v) {
      float val = D[v];
      if (val > best[v]) { best[v] = val; bestn[v] = cand; }
    }
  }
  // reduce across the 16 lanes of each half-wave; prefer lower index on ties
#pragma unroll
  for (int v = 0; v < 8; ++v) {
    float bv = best[v]; int bn = bestn[v];
#pragma unroll
    for (int m = 8; m; m >>= 1) {
      float ov = __shfl_xor(bv, m, 32);
      int   on = __shfl_xor(bn, m, 32);
      if (ov > bv || (ov == bv && on < bn)) { bv = ov; bn = on; }
    }
    if ((lane & 15) == 0) idx[r0 + ((lane >> 4) << 3) + v] = bn;
  }
}

// ------- embed gather + separable bicubic upsample to 32x32 (fused) -------
__global__ void k_upsample(const float* __restrict__ Wemb, const int* __restrict__ idx,
                           float* __restrict__ hup, int pn) {
  int e = blockIdx.x * 256 + threadIdx.x;
  int x = e & 31, y = (e >> 5) & 31, c = (e >> 10) & 31, b = e >> 15;
  float acc;
  if (pn == H_) {
    int code = idx[(b * H_ + y) * H_ + x];
    acc = Wemb[(size_t)code * C_ + c];
  } else {
    float scale = (float)pn * (1.0f / 32.0f);
    float sy = (y + 0.5f) * scale - 0.5f;
    float sx = (x + 0.5f) * scale - 0.5f;
    int iy0 = (int)floorf(sy), ix0 = (int)floorf(sx);
    float ty = sy - iy0, tx = sx - ix0;
    float wy[4], wx[4];
#pragma unroll
    for (int k = 0; k < 4; ++k) {
      wy[k] = cubw(ty - (float)(k - 1));
      wx[k] = cubw(tx - (float)(k - 1));
    }
    acc = 0.f;
#pragma unroll
    for (int ky = 0; ky < 4; ++ky) {
      int cy = iy0 + ky - 1; cy = cy < 0 ? 0 : (cy > pn - 1 ? pn - 1 : cy);
#pragma unroll
      for (int kx = 0; kx < 4; ++kx) {
        int cx = ix0 + kx - 1; cx = cx < 0 ? 0 : (cx > pn - 1 ? pn - 1 : cx);
        int code = idx[(b * pn + cy) * pn + cx];
        acc += wy[ky] * wx[kx] * Wemb[(size_t)code * C_ + c];
      }
    }
  }
  hup[e] = acc;
}

// ------- Phi (3x3 conv blend) + residual update + loss partial -------
__global__ void __launch_bounds__(256)
k_phi(const float* __restrict__ hup, const float* __restrict__ w,
      const float* __restrict__ bias, const float* __restrict__ f,
      float* __restrict__ fhat, float* __restrict__ frest,
      float* __restrict__ partial) {
  __shared__ float s_in[C_ * 10 * 32];   // 32 ch x 10 rows x 32 cols (40 KB)
  __shared__ float s_w[289];             // 32*3*3 weights for this o + bias
  __shared__ float s_red[256];
  int t  = threadIdx.x;
  int e0 = blockIdx.x * 256;
  int b  = e0 >> 15;
  int o  = (e0 >> 10) & 31;              // whole block shares one output channel
  int y0 = (e0 >> 5) & 31;               // multiple of 8

  if (t < 288) s_w[t] = w[o * 288 + t];
  if (t == 0)  s_w[288] = bias[o];
  for (int l = t; l < C_ * 10 * 32; l += 256) {
    int ch  = l / 320;
    int rem = l - ch * 320;
    int r   = rem >> 5;
    int xx  = rem & 31;
    int gy  = y0 + r - 1;
    s_in[l] = (gy >= 0 && gy < H_)
                  ? hup[(((size_t)b * C_ + ch) * H_ + gy) * H_ + xx] : 0.f;
  }
  __syncthreads();

  int ty = t >> 5, x = t & 31;
  float acc = 0.f;
  for (int i = 0; i < C_; ++i) {
    const float* ti = s_in + i * 320 + ty * 32;
#pragma unroll
    for (int dy = 0; dy < 3; ++dy) {
      const float* tr = ti + dy * 32;
      float w0 = s_w[i * 9 + dy * 3 + 0];
      float w1 = s_w[i * 9 + dy * 3 + 1];
      float w2 = s_w[i * 9 + dy * 3 + 2];
      if (x > 0)  acc += tr[x - 1] * w0;
      acc += tr[x] * w1;
      if (x < 31) acc += tr[x + 1] * w2;
    }
  }
  int e = e0 + t;
  float center = s_in[o * 320 + (ty + 1) * 32 + x];
  float ph = 0.5f * center + 0.5f * (acc + s_w[288]);   // RESI = 0.5
  float fh = fhat[e] + ph;
  fhat[e]  = fh;
  frest[e] -= ph;
  float d = fh - f[e];
  s_red[t] = d * d;
  __syncthreads();
  for (int s = 128; s; s >>= 1) {
    if (t < s) s_red[t] += s_red[t + s];
    __syncthreads();
  }
  if (t == 0) partial[blockIdx.x] = s_red[0];
}

__global__ void k_lred(const float* __restrict__ partial, float* __restrict__ loss) {
  __shared__ float s[256];
  int t = threadIdx.x;
  float a = 0.f;
  for (int i = t; i < 2048; i += 256) a += partial[i];
  s[t] = a; __syncthreads();
  for (int st = 128; st; st >>= 1) { if (t < st) s[t] += s[t + st]; __syncthreads(); }
  // loss += (1+BETA)/SN * mean  ==  1.25/(6*NELEM) * sumsq
  if (t == 0) loss[0] += s[0] * (1.25f / (6.0f * 524288.0f));
}

// ---------------- finalize: histogram, one-hot enc, idx, perplexity ----------
__global__ void k_count(const int* __restrict__ idx, int* __restrict__ counts) {
  int r = blockIdx.x * 256 + threadIdx.x;
  atomicAdd(&counts[idx[r]], 1);                 // integer: deterministic
}

__global__ void k_enc(const int* __restrict__ idx, float* __restrict__ enc,
                      float* __restrict__ idxf) {
  int r = blockIdx.x * 256 + threadIdx.x;
  int id = idx[r];
  enc[(size_t)r * NE_ + id] = 1.0f;
  idxf[r] = (float)id;
}

__global__ void k_perp(const int* __restrict__ counts, float* __restrict__ perp) {
  __shared__ float s[256];
  int t = threadIdx.x;
  float a = 0.f;
  for (int c = t; c < NE_; c += 256) {
    float p = (float)counts[c] * (1.0f / 16384.0f);
    a += p * logf(p + 1e-10f);
  }
  s[t] = a; __syncthreads();
  for (int st = 128; st; st >>= 1) { if (t < st) s[t] += s[t + st]; __syncthreads(); }
  if (t == 0) perp[0] = expf(-s[0]);
}

extern "C" void kernel_launch(void* const* d_in, const int* in_sizes, int n_in,
                              void* d_out, int out_size, void* d_ws, size_t ws_size,
                              hipStream_t stream) {
  (void)in_sizes; (void)n_in; (void)out_size; (void)ws_size;
  const float* f    = (const float*)d_in[0];   // [16,32,32,32]
  const float* Wemb = (const float*)d_in[1];   // [4096,32]
  const float* phiw = (const float*)d_in[2];   // [4,32,32,3,3]
  const float* phib = (const float*)d_in[3];   // [4,32]
  float* out = (float*)d_out;

  char* ws = (char*)d_ws;                      // ~5.4 MB used
  float*    frest   = (float*)(ws + 0);        // 2 MB
  float*    hup     = (float*)(ws + 2097152);  // 2 MB
  _Float16* rn      = (_Float16*)(ws + 4194304);   // 1 MB
  _Float16* Wn      = (_Float16*)(ws + 5242880);   // 256 KB
  int*      idxbuf  = (int*)(ws + 5505024);        // 64 KB
  int*      counts  = (int*)(ws + 5570560);        // 16 KB
  float*    partial = (float*)(ws + 5586944);      // 8 KB

  k_init<<<NELEM / 256, 256, 0, stream>>>(f, out, frest, counts);
  k_zero_enc<<<65536, 256, 0, stream>>>(out + ENC_OFF);
  k_normw<<<NE_ / 8, 256, 0, stream>>>(Wemb, Wn);

  static const int PN[6] = {1, 2, 4, 8, 16, 32};
  static const int PM[6] = {0, 0, 1, 2, 3, 3};   // PhiPartiallyShared map
  for (int si = 0; si < 6; ++si) {
    int pn = PN[si];
    int R  = B_ * pn * pn;
    k_prep<<<(R + 7) / 8, 256, 0, stream>>>(frest, rn, pn, R);
    int ntiles = R / 16;
    k_search<<<(ntiles + 3) / 4, 128, 0, stream>>>(rn, Wn, idxbuf, ntiles);
    k_upsample<<<NELEM / 256, 256, 0, stream>>>(Wemb, idxbuf, hup, pn);
    k_phi<<<NELEM / 256, 256, 0, stream>>>(hup, phiw + PM[si] * 288 * C_,
                                           phib + PM[si] * C_, f, out, frest,
                                           partial);
    k_lred<<<1, 256, 0, stream>>>(partial, out + LOSS_OFF);
  }
  k_count<<<64, 256, 0, stream>>>(idxbuf, counts);
  k_enc<<<64, 256, 0, stream>>>(idxbuf, out + ENC_OFF, out + IDX_OFF);
  k_perp<<<1, 256, 0, stream>>>(counts, out + PERP_OFF);
}
